// Online_SRNN_GLN_Cell_26061861552910
// MI455X (gfx1250) — matrix operand order repaired and checked by reading, not compiled, not run on hardware
//
#include <hip/hip_runtime.h>
#include <hip/hip_bf16.h>

typedef __attribute__((ext_vector_type(16))) __bf16 v16bf;
typedef __attribute__((ext_vector_type(8)))  float  v8f;

#define LOGIT_CLIP 5.0f

constexpr int B = 4, S = 512, P = 128, C = 64, K = 64, NBITS = 4, NCTX = 16;
constexpr int BS = B * S;       // 2048 tokens
constexpr int MT = BS / 16;     // 128 row tiles

constexpr int W1_N  = C * K * NCTX * P;   // 8388608
constexpr int H1_N  = C * K * NBITS * P;  // 2097152
constexpr int H2_N  = C * NBITS * P;      // 32768
constexpr int W2_N  = C * NCTX * K;       // 65536
constexpr int FC2_N = C * P;              // 8192
constexpr int X_N   = B * S * P;          // 262144

// workspace layout (bytes); every region is 32B aligned by construction
constexpr size_t OFF_BWS  = 0;
constexpr size_t OFF_W1B  = OFF_BWS  + (size_t)BS * C * 4;   //   512 KB
constexpr size_t OFF_H1B  = OFF_W1B  + (size_t)W1_N * 2;     // +16 MB
constexpr size_t OFF_H2B  = OFF_H1B  + (size_t)H1_N * 2;     // + 4 MB
constexpr size_t OFF_W2B  = OFF_H2B  + (size_t)H2_N * 2;
constexpr size_t OFF_FC2B = OFF_W2B  + (size_t)W2_N * 2;
constexpr size_t OFF_XRB  = OFF_FC2B + (size_t)FC2_N * 2;
constexpr size_t OFF_XCB  = OFF_XRB  + (size_t)X_N * 2;
constexpr size_t WS_NEEDED = OFF_XCB + (size_t)X_N * 2;      // ~21.7 MB

// ---- bf16 packing --------------------------------------------------------
__device__ __forceinline__ unsigned pack_bf16x2(float a, float b) {
  unsigned short ua = __builtin_bit_cast(unsigned short, (__bf16)a);
  unsigned short ub = __builtin_bit_cast(unsigned short, (__bf16)b);
  return (unsigned)ua | ((unsigned)ub << 16);
}

// ---- fragment loaders (A and B share contiguous k order, wave32) -------------
// A (16x32 bf16): lane l holds row M=l&15; element i -> K = (l>>4)*16 + i
template<bool CLIP>
__device__ __forceinline__ v16bf cvt_a_frag(const float* p0, const float* p1) {
  float4 f0 = ((const float4*)p0)[0];
  float4 f1 = ((const float4*)p0)[1];
  float4 f2 = ((const float4*)p1)[0];
  float4 f3 = ((const float4*)p1)[1];
  float t[16] = { f0.x,f0.y,f0.z,f0.w, f1.x,f1.y,f1.z,f1.w,
                  f2.x,f2.y,f2.z,f2.w, f3.x,f3.y,f3.z,f3.w };
  v16bf a;
#pragma unroll
  for (int i = 0; i < 16; ++i) {
    float v = t[i];
    if (CLIP) v = fminf(fmaxf(v, -LOGIT_CLIP), LOGIT_CLIP);
    a[i] = (__bf16)v;
  }
  return a;
}

__device__ __forceinline__ v16bf load_a_pre(const __bf16* p) {  // 8 bf16 @p, 8 @p+8
  union { uint4 u[2]; v16bf v; } un;
  un.u[0] = *(const uint4*)p;
  un.u[1] = *(const uint4*)(p + 8);
  return un.v;
}

// B (32x16 bf16): lane l holds column N=l&15; element i -> K = (l>>4)*16 + i
__device__ __forceinline__ v16bf cvt_b_frag(const float* p) {
  float4 f0 = ((const float4*)p)[0];
  float4 f1 = ((const float4*)p)[1];
  float4 f2 = ((const float4*)p)[2];
  float4 f3 = ((const float4*)p)[3];
  float t[16] = { f0.x,f0.y,f0.z,f0.w, f1.x,f1.y,f1.z,f1.w,
                  f2.x,f2.y,f2.z,f2.w, f3.x,f3.y,f3.z,f3.w };
  v16bf b;
#pragma unroll
  for (int i = 0; i < 16; ++i) b[i] = (__bf16)t[i];
  return b;
}

template<bool PRE>
__device__ __forceinline__ v16bf fetch_b_frag(const float* f32, const __bf16* b16, size_t off) {
  if constexpr (PRE) {
    union { uint4 u[2]; v16bf v; } un;
    un.u[0] = *(const uint4*)(b16 + off);
    un.u[1] = *(const uint4*)(b16 + off + 8);
    return un.v;
  } else {
    return cvt_b_frag(f32 + off);
  }
}

__device__ __forceinline__ float fast_sigmoid(float z) {
  return 1.0f / (1.0f + __expf(-z));
}

// ===========================================================================
// Kernel 0: fp32 -> bf16 streaming convert (optionally clipped), 8 elems/thread.
// ===========================================================================
__global__ __launch_bounds__(256)
void cvt_bf16_kernel(const float* __restrict__ src, unsigned* __restrict__ dst,
                     int n8, int clip) {
  int t = blockIdx.x * 256 + threadIdx.x;
  if (t >= n8) return;
  float4 a = ((const float4*)src)[t * 2];
  float4 b = ((const float4*)src)[t * 2 + 1];
  if (clip) {
    a.x = fminf(fmaxf(a.x, -LOGIT_CLIP), LOGIT_CLIP);
    a.y = fminf(fmaxf(a.y, -LOGIT_CLIP), LOGIT_CLIP);
    a.z = fminf(fmaxf(a.z, -LOGIT_CLIP), LOGIT_CLIP);
    a.w = fminf(fmaxf(a.w, -LOGIT_CLIP), LOGIT_CLIP);
    b.x = fminf(fmaxf(b.x, -LOGIT_CLIP), LOGIT_CLIP);
    b.y = fminf(fmaxf(b.y, -LOGIT_CLIP), LOGIT_CLIP);
    b.z = fminf(fmaxf(b.z, -LOGIT_CLIP), LOGIT_CLIP);
    b.w = fminf(fmaxf(b.w, -LOGIT_CLIP), LOGIT_CLIP);
  }
  uint4 o;
  o.x = pack_bf16x2(a.x, a.y);
  o.y = pack_bf16x2(a.z, a.w);
  o.z = pack_bf16x2(b.x, b.y);
  o.w = pack_bf16x2(b.z, b.w);
  ((uint4*)dst)[t] = o;
}

// ===========================================================================
// Kernel 1: per (16-token tile, class) block: GLN layer1 + layer2 + fc2 gate.
// One wave/block; every dot product is a v_wmma_f32_16x16x32_bf16.
// PRE=true: operands pre-converted to bf16 (2x b128/lane per WMMA k-step).
// ===========================================================================
template<bool PRE>
__global__ __launch_bounds__(32)
void gln_forward_kernel(const float* __restrict__ x,
                        const float* __restrict__ H1,
                        const float* __restrict__ W1,
                        const float* __restrict__ H2,
                        const float* __restrict__ W2,
                        const float* __restrict__ fc2w,
                        const float* __restrict__ fc2b,
                        const __bf16* __restrict__ xrawb,
                        const __bf16* __restrict__ xclipb,
                        const __bf16* __restrict__ H1b,
                        const __bf16* __restrict__ W1b,
                        const __bf16* __restrict__ H2b,
                        const __bf16* __restrict__ W2b,
                        const __bf16* __restrict__ fc2wb,
                        float* __restrict__ bws) {
  const int mt = blockIdx.x;     // token tile (16 rows of B*S)
  const int c  = blockIdx.y;     // class
  const int l  = threadIdx.x;    // lane
  const int n  = l & 15;         // fragment column / A row
  const int hi = l >> 4;         // lane half

  __shared__ float mixLds[NCTX * 16 * 16];  // [ctx][N][M] — b128 spill layout
  __shared__ float l1Lds[16 * K];           // [M][k] gathered+clipped layer-1 logits
  __shared__ float hdrLds[16 * 8];          // ctx2 bits (n=0..3) + fc2 logit (n=4)
  __shared__ float mix2Lds[16 * 16];        // [ctx][M]

  // ---- A fragments: raw (contexts) and clipped (mix) ----
  v16bf a_raw[4], a_clip[4];
#pragma unroll
  for (int ks = 0; ks < 4; ++ks) {
    const int aoff = (mt * 16 + n) * P + ks * 32 + (hi << 4);
    if constexpr (PRE) {
      a_raw[ks]  = load_a_pre(xrawb + aoff);
      a_clip[ks] = load_a_pre(xclipb + aoff);
    } else {
      a_raw[ks]  = cvt_a_frag<false>(x + aoff, x + aoff + 8);
      a_clip[ks] = cvt_a_frag<true >(x + aoff, x + aoff + 8);
    }
  }

  // ---- layer 1: loop over 4 neuron tiles of 16 ----
#pragma unroll 1
  for (int nt = 0; nt < 4; ++nt) {
    const int kneu = nt * 16 + n;  // this lane's neuron column

    // context-bit GEMMs: x . H1[c, kneu, j, :]
    v8f cacc[4];
#pragma unroll
    for (int j = 0; j < 4; ++j) {
      v8f acc = {};
#pragma unroll
      for (int ks = 0; ks < 4; ++ks) {
        size_t off = ((size_t)(c * K + kneu) * NBITS + j) * P + ks * 32 + (hi << 4);
        v16bf bf = fetch_b_frag<PRE>(H1, H1b, off);
        acc = __builtin_amdgcn_wmma_f32_16x16x32_bf16(false, a_raw[ks], false, bf,
                                                      (short)0, acc, false, false);
      }
      cacc[j] = acc;
    }
    unsigned ctxb[8];
#pragma unroll
    for (int r = 0; r < 8; ++r) {
      unsigned cb = 0;
      cb |= (cacc[0][r] > 0.0f) ? 1u : 0u;
      cb |= (cacc[1][r] > 0.0f) ? 2u : 0u;
      cb |= (cacc[2][r] > 0.0f) ? 4u : 0u;
      cb |= (cacc[3][r] > 0.0f) ? 8u : 0u;
      ctxb[r] = cb;
    }

    // mix GEMMs for all 16 contexts; spill each context tile to LDS (2x b128)
#pragma unroll 1
    for (int cx = 0; cx < NCTX; ++cx) {
      size_t wo = ((size_t)(c * K + kneu) * NCTX + cx) * P;
      if (cx + 1 < NCTX) {
        if constexpr (PRE) __builtin_prefetch(W1b + wo + P, 0, 1);
        else               __builtin_prefetch(W1 + wo + P, 0, 1);
      }
      v8f acc = {};
#pragma unroll
      for (int ks = 0; ks < 4; ++ks) {
        v16bf bf = fetch_b_frag<PRE>(W1, W1b, wo + ks * 32 + (hi << 4));
        acc = __builtin_amdgcn_wmma_f32_16x16x32_bf16(false, a_clip[ks], false, bf,
                                                      (short)0, acc, false, false);
      }
      float* dst = &mixLds[cx * 256 + n * 16 + (hi << 3)];
      *(float4*)(dst)     = make_float4(acc[0], acc[1], acc[2], acc[3]);
      *(float4*)(dst + 4) = make_float4(acc[4], acc[5], acc[6], acc[7]);
    }
    __syncthreads();

    // gather the selected context, clip, store l1 tile
#pragma unroll
    for (int r = 0; r < 8; ++r) {
      int mm = r + (hi << 3);
      float v = mixLds[ctxb[r] * 256 + n * 16 + mm];
      v = fminf(fmaxf(v, -LOGIT_CLIP), LOGIT_CLIP);
      l1Lds[mm * K + nt * 16 + n] = v;
    }
    __syncthreads();  // mixLds reused next neuron tile
  }

  // ---- layer 2 header GEMM: columns 0..3 = H2 bits, column 4 = fc2 gate ----
  {
    v8f acc = {};
#pragma unroll
    for (int ks = 0; ks < 4; ++ks) {
      v16bf bf;
      if (n < 4) {
        bf = fetch_b_frag<PRE>(H2, H2b, ((size_t)(c * NBITS + n)) * P + ks * 32 + (hi << 4));
      } else if (n == 4) {
        bf = fetch_b_frag<PRE>(fc2w, fc2wb, (size_t)c * P + ks * 32 + (hi << 4));
      } else {
        bf = v16bf{};
      }
      acc = __builtin_amdgcn_wmma_f32_16x16x32_bf16(false, a_raw[ks], false, bf,
                                                    (short)0, acc, false, false);
    }
    if (n < 5) {
#pragma unroll
      for (int r = 0; r < 8; ++r) hdrLds[(r + (hi << 3)) * 8 + n] = acc[r];
    }
  }

  // ---- layer 2 mix GEMM: l1 (16x64, from LDS) x W2[c, ctx, k] ----
  {
    v8f acc = {};
#pragma unroll
    for (int ks = 0; ks < 2; ++ks) {
      const float* ap = l1Lds + n * K + ks * 32 + (hi << 4);
      v16bf af = cvt_a_frag<false>(ap, ap + 8);
      v16bf bf = fetch_b_frag<PRE>(W2, W2b, ((size_t)(c * NCTX + n)) * K + ks * 32 + (hi << 4));
      acc = __builtin_amdgcn_wmma_f32_16x16x32_bf16(false, af, false, bf,
                                                    (short)0, acc, false, false);
    }
    float* dst = &mix2Lds[n * 16 + (hi << 3)];
    *(float4*)(dst)     = make_float4(acc[0], acc[1], acc[2], acc[3]);
    *(float4*)(dst + 4) = make_float4(acc[4], acc[5], acc[6], acc[7]);
  }
  __syncthreads();

  // ---- gather ctx2, double sigmoid, emit b[token, c] ----
  if (l < 16) {
    int mm = l;
    unsigned cx = 0;
    cx |= (hdrLds[mm * 8 + 0] > 0.0f) ? 1u : 0u;
    cx |= (hdrLds[mm * 8 + 1] > 0.0f) ? 2u : 0u;
    cx |= (hdrLds[mm * 8 + 2] > 0.0f) ? 4u : 0u;
    cx |= (hdrLds[mm * 8 + 3] > 0.0f) ? 8u : 0u;
    float alpha = hdrLds[mm * 8 + 4] + fc2b[c];
    float l2 = mix2Lds[cx * 16 + mm];
    bws[(size_t)(mt * 16 + mm) * C + c] = fast_sigmoid(l2) * fast_sigmoid(alpha);
  }
}

// ===========================================================================
// Kernel 2: sequential roll+add+relu scan. One wave per batch, 2 ch/lane,
// roll via __shfl, next-step load software-pipelined past the serial chain.
// ===========================================================================
__global__ __launch_bounds__(32)
void scan_kernel(const float* __restrict__ bws,
                 const float* __restrict__ hidden,
                 float* __restrict__ out) {
  const int bat = blockIdx.x;
  const int l = threadIdx.x;
  const int src = (l + 31) & 31;  // lane - 1 (mod 32)

  float h0 = hidden[bat * C + l];
  float h1 = hidden[bat * C + 32 + l];
  float* hlast = out + (size_t)B * S * C;

  int base = (bat * S) * C;
  float nb0 = bws[base + l];
  float nb1 = bws[base + 32 + l];

  for (int s = 0; s < S; ++s) {
    float b0 = nb0, b1 = nb1;
    if (s + 1 < S) {                       // prefetch next step off the chain
      nb0 = bws[base + C + l];
      nb1 = bws[base + C + 32 + l];
    }
    float p0 = __shfl(h0, src, 32);
    float p1 = __shfl(h1, src, 32);
    float r0 = (l == 0) ? p1 : p0;         // c=0 wraps to c=63 (hi of lane 31)
    float r1 = (l == 0) ? p0 : p1;         // c=32 takes c=31 (lo of lane 31)
    h0 = fmaxf(b0 + r0, 0.0f);
    h1 = fmaxf(b1 + r1, 0.0f);
    out[base + l] = h0;
    out[base + 32 + l] = h1;
    base += C;
  }
  hlast[bat * C + l] = h0;
  hlast[bat * C + 32 + l] = h1;
}

// ===========================================================================
extern "C" void kernel_launch(void* const* d_in, const int* in_sizes, int n_in,
                              void* d_out, int out_size, void* d_ws, size_t ws_size,
                              hipStream_t stream) {
  const float* x    = (const float*)d_in[0];
  const float* hid  = (const float*)d_in[1];
  const float* H1   = (const float*)d_in[2];
  const float* W1   = (const float*)d_in[3];
  const float* H2   = (const float*)d_in[4];
  const float* W2   = (const float*)d_in[5];
  const float* fc2w = (const float*)d_in[6];
  const float* fc2b = (const float*)d_in[7];
  float* out = (float*)d_out;

  char* ws = (char*)d_ws;
  float*  bws    = (float*)(ws + OFF_BWS);
  __bf16* W1b    = (__bf16*)(ws + OFF_W1B);
  __bf16* H1b    = (__bf16*)(ws + OFF_H1B);
  __bf16* H2b    = (__bf16*)(ws + OFF_H2B);
  __bf16* W2b    = (__bf16*)(ws + OFF_W2B);
  __bf16* fc2wb  = (__bf16*)(ws + OFF_FC2B);
  __bf16* xrawb  = (__bf16*)(ws + OFF_XRB);
  __bf16* xclipb = (__bf16*)(ws + OFF_XCB);

  const bool pre = (ws_size >= WS_NEEDED);
  dim3 g1(MT, C);

  if (pre) {
    auto cv = [&](const float* s, __bf16* d, int nelem, int clip) {
      int n8 = nelem / 8;
      cvt_bf16_kernel<<<(n8 + 255) / 256, 256, 0, stream>>>(s, (unsigned*)d, n8, clip);
    };
    cv(W1,   W1b,   W1_N,  0);
    cv(H1,   H1b,   H1_N,  0);
    cv(H2,   H2b,   H2_N,  0);
    cv(W2,   W2b,   W2_N,  0);
    cv(fc2w, fc2wb, FC2_N, 0);
    cv(x,    xrawb, X_N,   0);
    cv(x,    xclipb, X_N,  1);
    gln_forward_kernel<true><<<g1, 32, 0, stream>>>(x, H1, W1, H2, W2, fc2w, fc2b,
                                                    xrawb, xclipb, H1b, W1b, H2b, W2b,
                                                    fc2wb, bws);
  } else {
    gln_forward_kernel<false><<<g1, 32, 0, stream>>>(x, H1, W1, H2, W2, fc2w, fc2b,
                                                     nullptr, nullptr, nullptr, nullptr,
                                                     nullptr, nullptr, nullptr, bws);
  }
  scan_kernel<<<B, 32, 0, stream>>>(bws, hid, out);
}
